// TopKFrozenEmbeddings_29953101923041
// MI455X (gfx1250) — matrix operand order repaired and checked by reading, not compiled, not run on hardware
//
#include <hip/hip_runtime.h>
#include <hip/hip_bf16.h>

typedef __attribute__((ext_vector_type(16))) __bf16 v16bf;
typedef __attribute__((ext_vector_type(8)))  __bf16 v8bf;
typedef __attribute__((ext_vector_type(8)))  float  v8f;

constexpr int kH    = 768;   // full hidden dim
constexpr int kHR   = 192;   // reduced dim (H / R)
constexpr int kTopK = 10;
constexpr int kPadRows = 192;           // OOB-safe pad rows on packed emb table
constexpr float kLow = -1.0e30f;        // finite sentinel (avoids inf-inf NaN)

// ---------------------------------------------------------------------------
// Phase 0: pack x[:, ::4]*2 and emb[:, ::4]*2 into bf16 workspace arrays.
// ---------------------------------------------------------------------------
__global__ void pack_kernel(const float* __restrict__ x, const float* __restrict__ e,
                            __bf16* __restrict__ xr, __bf16* __restrict__ er,
                            int N, int V) {
  long idx = (long)blockIdx.x * blockDim.x + threadIdx.x;
  long nx = (long)N * kHR;
  long total = nx + (long)V * kHR;
  if (idx >= total) return;
  if (idx < nx) {
    long n = idx / kHR, j = idx % kHR;
    xr[idx] = (__bf16)(2.0f * x[n * kH + j * 4]);
  } else {
    long t = idx - nx;
    long v = t / kHR, j = t % kHR;
    er[t] = (__bf16)(2.0f * e[v * kH + j * 4]);
  }
}

// Load one 16-column B tile (32x16 bf16 tile in contiguous k order, as A: lane = N column,
// lanes 16..31 carry K 16..31 of the same column), via 32-bit byte offset.
__device__ __forceinline__ void loadB(v16bf (&b)[6], const char* __restrict__ er8,
                                      unsigned ofs) {
  #pragma unroll
  for (int c = 0; c < 6; ++c)
    b[c] = *(const v16bf*)(er8 + ofs + c * 64);
}

// ---------------------------------------------------------------------------
// Phase 1+2 fused.
// ---------------------------------------------------------------------------
__launch_bounds__(128)
__global__ void fused_kernel(const float* __restrict__ xf, const float* __restrict__ ef,
                             const __bf16* __restrict__ xr, const __bf16* __restrict__ er,
                             float* __restrict__ out, int N, int V) {
  __shared__ float tileS[4][16][17];          // per-wave C-tile buffer (slow path)
  __shared__ float mM[4][16], mS[4][16];      // per-wave online-softmax state
  __shared__ float mTV[4][16][kTopK];
  __shared__ int   mTI[4][16][kTopK];
  __shared__ float fSc[16][kTopK];            // merged approx softmax scores
  __shared__ int   fIx[16][kTopK];            // merged top-10 indices

  const int tid  = threadIdx.x;
  const int w    = tid >> 5;                  // wave id 0..3 (wave32)
  const int lane = tid & 31;
  const int m16  = lane & 15;
  const int hi   = lane >> 4;                 // half-wave selector
  const int row0 = blockIdx.x * 16;

  // ---- A tile (x_red, 16 rows x 192) resident in registers.
  v16bf a[6];
  {
    const __bf16* xrow = xr + (long)(row0 + m16) * kHR;
    #pragma unroll
    for (int c = 0; c < 6; ++c) {
      v8bf lo = *(const v8bf*)(xrow + c * 32 + hi * 16);
      v8bf h2 = *(const v8bf*)(xrow + c * 32 + hi * 16 + 8);
      #pragma unroll
      for (int i = 0; i < 8; ++i) { a[c][i] = lo[i]; a[c][i + 8] = h2[i]; }
    }
  }

  // Per-lane online softmax state: VGPR r of the C tile carries row (r + hi*8);
  // each lane tracks its own (max, sum) over the columns it sees; merged once
  // at the end with a log-sum-exp shuffle tree.
  float pm[8], ps[8], thr[8];
  #pragma unroll
  for (int r = 0; r < 8; ++r) { pm[r] = kLow; ps[r] = 0.0f; thr[r] = kLow; }

  // Top-10 lists: lane m (< 16) owns row m.
  float tv[kTopK]; int ti[kTopK];
  #pragma unroll
  for (int q = 0; q < kTopK; ++q) { tv[q] = kLow; ti[q] = 0; }
  float tmin = kLow; int tms = 0;

  auto processTile = [&](const v16bf (&b)[6], int cb) {
    v8f acc = {};
    #pragma unroll
    for (int c = 0; c < 6; ++c)
      acc = __builtin_amdgcn_wmma_f32_16x16x32_bf16(false, a[c], false, b[c],
                                                    (short)0, acc, false, false);
    // Stash tile for the (rare) top-k slow path.
    #pragma unroll
    for (int r = 0; r < 8; ++r)
      tileS[w][r + hi * 8][m16] = acc[r];

    const bool valid = (cb + m16) < V;
    unsigned flags = 0;
    #pragma unroll
    for (int r = 0; r < 8; ++r) {
      float v = valid ? acc[r] : kLow;
      // branchless per-lane online softmax update, single exp:
      float d = v - pm[r];
      float e = __expf(-fabsf(d));
      bool gt = d > 0.0f;
      ps[r] = gt ? fmaf(ps[r], e, 1.0f) : (ps[r] + e);
      pm[r] = gt ? v : pm[r];
      // top-k candidate detection: wave-uniform ballot, scalar decode.
      unsigned bm = (unsigned)__ballot(v > thr[r]);
      if (bm & 0xFFFFu) flags |= 1u << r;        // row r       (lanes 0..15)
      if (bm >> 16)     flags |= 1u << (r + 8);  // row r + 8   (lanes 16..31)
    }
    if (flags) {                                 // scalar-uniform rare path
      if (lane < 16 && ((flags >> lane) & 1u)) {
        #pragma unroll 4
        for (int j = 0; j < 16; ++j) {
          int col = cb + j;
          float val = tileS[w][lane][j];
          if (col < V && val > tmin) {
            #pragma unroll
            for (int q = 0; q < kTopK; ++q)
              if (q == tms) { tv[q] = val; ti[q] = col; }
            tmin = tv[0]; tms = 0;
            #pragma unroll
            for (int q = 1; q < kTopK; ++q)
              if (tv[q] < tmin) { tmin = tv[q]; tms = q; }
          }
        }
      }
      // Re-broadcast thresholds (row r+hi*8 lives in lane r+hi*8's tmin).
      #pragma unroll
      for (int r = 0; r < 8; ++r)
        thr[r] = __shfl(tmin, r + hi * 8, 32);
    }
  };

  // ---- main V loop: 2-tile unrolled, register double-buffered B,
  // incremental 32-bit byte offsets (emb table is padded, no clamping).
  const char* er8 = (const char*)er;
  constexpr unsigned kStep = 64u * kHR * 2u;   // 24576 B per 64 columns
  unsigned ofs = (unsigned)((w * 16 + m16) * (kHR * 2)) + (unsigned)(hi * 32);

  v16bf b0[6], b1[6];
  loadB(b0, er8, ofs);
  for (int cb = w * 16; cb < V; cb += 128) {
    loadB(b1, er8, ofs + kStep);
    __builtin_prefetch(er8 + ofs + 2u * kStep, 0, 1);   // global_prefetch_b8
    processTile(b0, cb);
    loadB(b0, er8, ofs + 2u * kStep);
    if (cb + 64 < V) processTile(b1, cb + 64);
    ofs += 2u * kStep;
  }

  // ---- merge the 16 per-lane partial (max,sum) pairs per row (one-time).
  #pragma unroll
  for (int r = 0; r < 8; ++r) {
    float m = pm[r], s = ps[r];
    #pragma unroll
    for (int off = 1; off < 16; off <<= 1) {
      float mo = __shfl_xor(m, off, 32);
      float so = __shfl_xor(s, off, 32);
      float M = fmaxf(m, mo);
      s = s * __expf(m - M) + so * __expf(mo - M);
      m = M;
    }
    pm[r] = m; ps[r] = s;
  }

  // ---- publish per-wave state.
  if (m16 == 0) {
    #pragma unroll
    for (int r = 0; r < 8; ++r) { mM[w][hi * 8 + r] = pm[r]; mS[w][hi * 8 + r] = ps[r]; }
  }
  if (lane < 16) {
    #pragma unroll
    for (int q = 0; q < kTopK; ++q) { mTV[w][lane][q] = tv[q]; mTI[w][lane][q] = ti[q]; }
  }
  __syncthreads();

  // ---- merge the 4 wave stripes (threads 0..15, one row each).
  if (tid < 16) {
    const int m = tid;
    float M = mM[0][m];
    for (int ww = 1; ww < 4; ++ww) M = fmaxf(M, mM[ww][m]);
    float S = 0.0f;
    for (int ww = 0; ww < 4; ++ww) S += mS[ww][m] * __expf(mM[ww][m] - M);

    float btv[kTopK]; int bti[kTopK];
    #pragma unroll
    for (int q = 0; q < kTopK; ++q) { btv[q] = mTV[0][m][q]; bti[q] = mTI[0][m][q]; }
    float bmin = btv[0]; int bms = 0;
    #pragma unroll
    for (int q = 1; q < kTopK; ++q) if (btv[q] < bmin) { bmin = btv[q]; bms = q; }

    for (int ww = 1; ww < 4; ++ww) {
      for (int q2 = 0; q2 < kTopK; ++q2) {
        float val = mTV[ww][m][q2]; int col = mTI[ww][m][q2];
        if (val > bmin) {
          #pragma unroll
          for (int q = 0; q < kTopK; ++q)
            if (q == bms) { btv[q] = val; bti[q] = col; }
          bmin = btv[0]; bms = 0;
          #pragma unroll
          for (int q = 1; q < kTopK; ++q)
            if (btv[q] < bmin) { bmin = btv[q]; bms = q; }
        }
      }
    }
    #pragma unroll
    for (int q = 0; q < kTopK; ++q) {
      fSc[m][q] = __expf(btv[q] - M) / S;   // approx softmax top_scores
      fIx[m][q] = bti[q];
    }
  }
  __syncthreads();

  // ---- exact f32 rescore: wave w handles rows 4w..4w+3 of the tile.
  for (int rr = 0; rr < 4; ++rr) {
    const int m = w * 4 + rr;
    const int n = row0 + m;
    if (n >= N) continue;
    float xc[24];
    const float* xp = xf + (long)n * kH;
    #pragma unroll
    for (int j = 0; j < 24; ++j) xc[j] = xp[lane + 32 * j];

    float lg[kTopK];
    float lmax = kLow;
    #pragma unroll
    for (int q = 0; q < kTopK; ++q) {
      const float* ep = ef + (long)fIx[m][q] * kH;
      float p = 0.0f;
      #pragma unroll
      for (int j = 0; j < 24; ++j) p = fmaf(xc[j], ep[lane + 32 * j], p);
      #pragma unroll
      for (int off = 16; off > 0; off >>= 1) p += __shfl_xor(p, off, 32);
      lg[q] = p;
      lmax = fmaxf(lmax, p);
    }
    float ssum = 0.0f;
    #pragma unroll
    for (int q = 0; q < kTopK; ++q) { lg[q] = __expf(lg[q] - lmax); ssum += lg[q]; }
    float best = kLow;
    #pragma unroll
    for (int q = 0; q < kTopK; ++q)
      best = fmaxf(best, 0.5f * (lg[q] / ssum + fSc[m][q]));
    if (lane == 0) out[n] = best;
  }
}

// ---------------------------------------------------------------------------
extern "C" void kernel_launch(void* const* d_in, const int* in_sizes, int n_in,
                              void* d_out, int out_size, void* d_ws, size_t ws_size,
                              hipStream_t stream) {
  const float* xf = (const float*)d_in[0];   // [B,S,H] f32
  const float* ef = (const float*)d_in[1];   // [V,H]   f32
  const int N = in_sizes[0] / kH;            // 4096
  const int V = in_sizes[1] / kH;            // 50257

  char* ws = (char*)d_ws;
  __bf16* xr = (__bf16*)ws;
  size_t xrBytes = ((size_t)N * kHR * sizeof(__bf16) + 255) & ~(size_t)255;
  __bf16* er = (__bf16*)(ws + xrBytes);
  // er is sized for V + kPadRows rows; pad contents are never used unmasked.

  long total = (long)N * kHR + (long)V * kHR;
  int pThreads = 256;
  unsigned pBlocks = (unsigned)((total + pThreads - 1) / pThreads);
  pack_kernel<<<dim3(pBlocks), dim3(pThreads), 0, stream>>>(xf, ef, xr, er, N, V);

  unsigned rowTiles = (unsigned)((N + 15) / 16);
  fused_kernel<<<dim3(rowTiles), dim3(128), 0, stream>>>(xf, ef, xr, er,
                                                         (float*)d_out, N, V);
}
